// Adiin_24197845746021
// MI455X (gfx1250) — matrix operand order repaired and checked by reading, not compiled, not run on hardware
//
#include <hip/hip_runtime.h>

// ---------------------------------------------------------------------------
// CDNA5 (gfx1250) implementation of the SDCN-style forward pass.
// Large GEMMs: v_wmma_f32_16x16x32_bf16 (wave32, f32 accumulate), 128x64
// block tiles. B tiles are async-copied (global_load_async_to_lds_b128,
// ASYNCcnt, double-buffered) as raw fp32 into LDS, converted to bf16 ONCE
// per block per tile, then consumed as packed-bf16 ds_load_b128 fragments
// with zero per-wave conversion VALU. Gating MLPs / epilogues in fp32 VALU.
// ---------------------------------------------------------------------------

typedef __attribute__((ext_vector_type(16))) __bf16 v16bf;
typedef __attribute__((ext_vector_type(8)))  float  v8f;

union BF16x16 { v16bf v; unsigned short u[16]; uint4 q[2]; };

__device__ __forceinline__ unsigned short f2bf(float f) {
  // round-to-nearest-even fp32 -> bf16
  unsigned int x = __float_as_uint(f);
  unsigned int r = x + 0x7fffu + ((x >> 16) & 1u);
  return (unsigned short)(r >> 16);
}

#define RAWTILE 2048   // 32(K) x 64(N) fp32 elements per staged tile
#define HPITCH  40     // bf16 tile pitch in halfs: [64 n][40 k], 80B rows (16B aligned)

// ---------------------------------------------------------------------------
// Async staging of one full 32(K) x 64(N) fp32 B tile into LDS scratch.
// b128 copies: 512 x 16B / 32 lanes -> 16 per block, 2 per wave. ASYNCcnt.
// Raw layouts: NN -> Sraw[k*64 + n]; BT -> Sraw[n*32 + k].
// ---------------------------------------------------------------------------
template <bool BT>
__device__ __forceinline__ void stage_async(const float* __restrict__ B, int K, int Nc,
                                            int kt, int nb, float* bufp, int wav, int lane)
{
  unsigned ldsbase = (unsigned)(unsigned long long)(void*)bufp; // low 32b = LDS offset
#pragma unroll
  for (int i = 0; i < 2; ++i) {
    int v = wav * 64 + i * 32 + lane;          // 0..511 vec4 index
    unsigned long long goff; unsigned loff;
    if constexpr (BT) {
      int n = v >> 3, k0 = (v & 7) * 4;
      goff = (unsigned long long)(nb + n) * K + (unsigned)(kt + k0);
      loff = (unsigned)((n * 32 + k0) * 4);
    } else {
      int k = v >> 4, n0 = (v & 15) * 4;
      goff = (unsigned long long)(kt + k) * Nc + (unsigned)(nb + n0);
      loff = (unsigned)((k * 64 + n0) * 4);
    }
    unsigned long long ga = (unsigned long long)(const void*)(B + goff);
    unsigned la = ldsbase + loff;
    asm volatile("global_load_async_to_lds_b128 %0, %1, off"
                 :: "v"(la), "v"(ga) : "memory");
  }
}

// Branchless fallback staging for edge tiles (K tail / partial N): clamp the
// index into bounds, then select 0 (v_min + v_cndmask, no exec branches).
template <bool BT>
__device__ __forceinline__ void stage_sync(const float* __restrict__ B, int K, int Nc,
                                           int kt, int nb, float* bufp, int tid)
{
  for (int l = tid; l < RAWTILE; l += 256) {
    int n, k;
    if constexpr (BT) { n = l >> 5; k = l & 31; }
    else              { k = l >> 6; n = l & 63; }
    int gk = kt + k, gn = nb + n;
    int ck = min(gk, K - 1), cn = min(gn, Nc - 1);
    float v;
    if constexpr (BT) v = B[(unsigned long long)cn * K + ck];
    else              v = B[(unsigned long long)ck * Nc + cn];
    v = (gk < K && gn < Nc) ? v : 0.f;
    if constexpr (BT) bufp[n * 32 + k] = v;
    else              bufp[k * 64 + n] = v;
  }
}

// ---------------------------------------------------------------------------
// WMMA GEMM:  C[4096 x Nc] = act( A[4096 x K] * B + bias ),  fp32 row-major.
// BT=false: B is [K x Nc];  BT=true: B is [Nc x K] used transposed (NT GEMM).
// Block: 256 threads = 8 waves; tile 128(M) x 64(N), K-step 32; each wave
// owns a 16-row strip and 4 accumulators (4 WMMAs / K-step from 1 A frag).
// act: 0=none 1=relu 2=sigmoid.
//
// Fragment layouts per CDNA5 ISA 7.12.2 (wave32):
//   A 16x32 bf16 : lane row = lane%16, hf = lane/16; element e -> K =
//                  (e<8 ? hf*8+e : 16+hf*8+(e-8))  (two contiguous 8-runs)
//   B 32x16 bf16 : lane col = lane%16; element e -> K same as A (two 8-runs,
//                  exactly two packed-bf16 ds_load_b128 from Bh)
//   C/D 16x16 f32: acc[i] -> row hf*8+i, col lane%16
// ---------------------------------------------------------------------------
template <bool BT>
__global__ __launch_bounds__(256)
void gemm_wmma_kernel(const float* __restrict__ A, const float* __restrict__ B,
                      const float* __restrict__ bias, float* __restrict__ C,
                      int K, int Nc, int act)
{
  __shared__ __align__(16) float          Sraw[2 * RAWTILE];   // raw fp32 tiles (dbl buf)
  __shared__ __align__(16) unsigned short Bh[64 * HPITCH];     // bf16 tile [n][k]

  const int tid  = threadIdx.x;
  const int lane = tid & 31;
  const int wav  = tid >> 5;       // 0..7 -> M strip
  const int hf   = lane >> 4;      // 0..1
  const int l16  = lane & 15;

  const int m0 = blockIdx.y * 128 + wav * 16;
  const int nb = blockIdx.x * 64;
  const bool fullN = (nb + 64 <= Nc);

  const float* Arow = A + (unsigned long long)(m0 + l16) * K;

  v8f acc[4] = {};

  const int numT = (K + 31) >> 5;

  // prologue: stage tile 0
  if (fullN && 32 <= K) stage_async<BT>(B, K, Nc, 0, nb, Sraw, wav, lane);
  else                  stage_sync<BT>(B, K, Nc, 0, nb, Sraw, tid);

  for (int it = 0; it < numT; ++it) {
    const int kt = it << 5;
    float* cur = Sraw + ((it & 1) ? RAWTILE : 0);
    float* nxt = Sraw + ((it & 1) ? 0 : RAWTILE);

    // finish staging of current tile (own wave's asyncs), make LDS visible
    asm volatile("s_wait_asynccnt 0x0" ::: "memory");
    __syncthreads();

    // convert raw fp32 tile -> packed bf16 tile ONCE per block (8 elem/thread)
    for (int l = tid; l < RAWTILE; l += 256) {
      int n, k; float v;
      if constexpr (BT) { n = l >> 5; k = l & 31; v = cur[n * 32 + k]; }
      else              { k = l >> 6; n = l & 63; v = cur[k * 64 + n]; }
      Bh[n * HPITCH + k] = f2bf(v);
    }
    __syncthreads();

    // prefetch next tile into the other raw buffer while we compute
    if (it + 1 < numT) {
      const int kn = kt + 32;
      if (fullN && kn + 32 <= K) stage_async<BT>(B, K, Nc, kn, nb, nxt, wav, lane);
      else                       stage_sync<BT>(B, K, Nc, kn, nb, nxt, tid);
    }

    // ---------------- A fragment ----------------
    BF16x16 a;
    if (kt + 32 <= K) {
      const float* p0 = Arow + kt + (hf << 3);
      float2 t0 = *(const float2*)(p0 + 0);
      float2 t1 = *(const float2*)(p0 + 2);
      float2 t2 = *(const float2*)(p0 + 4);
      float2 t3 = *(const float2*)(p0 + 6);
      float2 t4 = *(const float2*)(p0 + 16);
      float2 t5 = *(const float2*)(p0 + 18);
      float2 t6 = *(const float2*)(p0 + 20);
      float2 t7 = *(const float2*)(p0 + 22);
      a.u[0]  = f2bf(t0.x); a.u[1]  = f2bf(t0.y);
      a.u[2]  = f2bf(t1.x); a.u[3]  = f2bf(t1.y);
      a.u[4]  = f2bf(t2.x); a.u[5]  = f2bf(t2.y);
      a.u[6]  = f2bf(t3.x); a.u[7]  = f2bf(t3.y);
      a.u[8]  = f2bf(t4.x); a.u[9]  = f2bf(t4.y);
      a.u[10] = f2bf(t5.x); a.u[11] = f2bf(t5.y);
      a.u[12] = f2bf(t6.x); a.u[13] = f2bf(t6.y);
      a.u[14] = f2bf(t7.x); a.u[15] = f2bf(t7.y);
      __builtin_prefetch(p0 + 32, 0, 0);   // next K tile (adj stays L2-resident)
    } else {
#pragma unroll
      for (int e = 0; e < 16; ++e) {
        int k = kt + ((e < 8) ? ((hf << 3) + e) : (8 + (hf << 3) + e));
        float v = Arow[min(k, K - 1)];               // in-bounds clamp
        a.u[e] = (k < K) ? f2bf(v) : (unsigned short)0;  // branchless zero-fill
      }
    }

    // ---------------- 4 x (bf16 B fragment + WMMA) ----------------
#pragma unroll
    for (int wn2 = 0; wn2 < 4; ++wn2) {
      const uint4* bp = (const uint4*)&Bh[(wn2 * 16 + l16) * HPITCH + (hf << 3)];
      BF16x16 b;
      b.q[0] = bp[0];            // 2x ds_load_b128, already packed bf16
      b.q[1] = bp[2];
      acc[wn2] = __builtin_amdgcn_wmma_f32_16x16x32_bf16(
          /*neg_a=*/false, a.v, /*neg_b=*/false, b.v,
          /*c_mod=*/(short)0, acc[wn2], /*reuse_a=*/false, /*reuse_b=*/false);
    }
  }

  // ---------------- epilogue ----------------
#pragma unroll
  for (int wn2 = 0; wn2 < 4; ++wn2) {
    const int cn = nb + wn2 * 16 + l16;
    if (cn < Nc) {
      float bv = bias ? bias[cn] : 0.f;
#pragma unroll
      for (int i = 0; i < 8; ++i) {
        int cm = m0 + (hf << 3) + i;
        float v = acc[wn2][i] + bv;
        if (act == 1)      v = fmaxf(v, 0.f);
        else if (act == 2) v = 1.f / (1.f + __expf(-v));
        C[(unsigned long long)cm * Nc + cn] = v;
      }
    }
  }
}

// ---------------------------------------------------------------------------
// Gate: P = l2norm(softmax(leaky_relu(cat(s0,s1,s2,s3) @ W + bias, 0.01)))
// One wave32 per row; NO <= 4 outputs. Pure fp32.
// ---------------------------------------------------------------------------
__global__ __launch_bounds__(256)
void gate_kernel(const float* __restrict__ s0, int w0,
                 const float* __restrict__ s1, int w1,
                 const float* __restrict__ s2, int w2,
                 const float* __restrict__ s3, int w3,
                 const float* __restrict__ W, const float* __restrict__ bias,
                 float* __restrict__ P, int NO)
{
  const int lane = threadIdx.x & 31;
  const int row  = blockIdx.x * 8 + (threadIdx.x >> 5);
  const int Ktot = w0 + w1 + w2 + w3;

  float accv[4] = {0.f, 0.f, 0.f, 0.f};
  for (int k = lane; k < Ktot; k += 32) {
    float xv;
    int kk = k;
    if (kk < w0) xv = s0[(size_t)row * w0 + kk];
    else {
      kk -= w0;
      if (kk < w1) xv = s1[(size_t)row * w1 + kk];
      else {
        kk -= w1;
        if (kk < w2) xv = s2[(size_t)row * w2 + kk];
        else { kk -= w2; xv = s3[(size_t)row * w3 + kk]; }
      }
    }
#pragma unroll
    for (int j = 0; j < 4; ++j)
      if (j < NO) accv[j] += xv * W[(size_t)k * NO + j];
  }
#pragma unroll
  for (int j = 0; j < 4; ++j)
#pragma unroll
    for (int off = 16; off > 0; off >>= 1)
      accv[j] += __shfl_down(accv[j], off, 32);

  if (lane == 0) {
    float v[4];
    float mx = -1e30f;
    for (int j = 0; j < NO; ++j) {
      float t = accv[j] + bias[j];
      t = (t > 0.f) ? t : 0.01f * t;       // leaky_relu
      v[j] = t; mx = fmaxf(mx, t);
    }
    float s = 0.f;
    for (int j = 0; j < NO; ++j) { v[j] = __expf(v[j] - mx); s += v[j]; }
    float n2 = 0.f;
    for (int j = 0; j < NO; ++j) { v[j] /= s; n2 += v[j] * v[j]; }
    float inv = 1.f / fmaxf(sqrtf(n2), 1e-12f);
    for (int j = 0; j < NO; ++j) P[(size_t)row * NO + j] = v[j] * inv;
  }
}

// O = P[:,0]*A + P[:,1]*B (row-wise scalars, width Wd)
__global__ void mix2_kernel(const float* __restrict__ P, const float* __restrict__ A,
                            const float* __restrict__ B, float* __restrict__ O,
                            int Wd, int total)
{
  int i = blockIdx.x * 256 + threadIdx.x;
  if (i >= total) return;
  int row = i / Wd;
  O[i] = P[row * 2 + 0] * A[i] + P[row * 2 + 1] * B[i];
}

__global__ void add_kernel(const float* __restrict__ A, const float* __restrict__ B,
                           float* __restrict__ O, int total)
{
  int i = blockIdx.x * 256 + threadIdx.x;
  if (i >= total) return;
  O[i] = A[i] + B[i];
}

// net = cat(w0*z1, w1*z2, w2*z3, w3*z)   widths 500,500,500,10 (total 1510)
__global__ void build_net_kernel(const float* __restrict__ W4,
                                 const float* __restrict__ z1, const float* __restrict__ z2,
                                 const float* __restrict__ z3, const float* __restrict__ z,
                                 float* __restrict__ net, int total)
{
  int i = blockIdx.x * 256 + threadIdx.x;
  if (i >= total) return;
  int row = i / 1510, c = i - row * 1510;
  float v;
  if (c < 500)       v = W4[row * 4 + 0] * z1[(size_t)row * 500 + c];
  else if (c < 1000) v = W4[row * 4 + 1] * z2[(size_t)row * 500 + (c - 500)];
  else if (c < 1500) v = W4[row * 4 + 2] * z3[(size_t)row * 500 + (c - 1000)];
  else               v = W4[row * 4 + 3] * z [(size_t)row * 10  + (c - 1500)];
  net[i] = v;
}

// row softmax, Nc <= 16
__global__ void softmax_rows_kernel(const float* __restrict__ X, float* __restrict__ O, int Nc)
{
  int row = blockIdx.x * 256 + threadIdx.x;
  if (row >= 4096) return;
  float v[16];
  float mx = -1e30f;
  for (int j = 0; j < Nc; ++j) { v[j] = X[(size_t)row * Nc + j]; mx = fmaxf(mx, v[j]); }
  float s = 0.f;
  for (int j = 0; j < Nc; ++j) { v[j] = __expf(v[j] - mx); s += v[j]; }
  for (int j = 0; j < Nc; ++j) O[(size_t)row * Nc + j] = v[j] / s;
}

// student-t soft assignment, v=1:  q_j = (1/(1+d2_j)) / sum
__global__ void studentt_kernel(const float* __restrict__ Z, const float* __restrict__ MU,
                                float* __restrict__ Q)
{
  int row = blockIdx.x * 256 + threadIdx.x;
  if (row >= 4096) return;
  float zz[10];
#pragma unroll
  for (int d = 0; d < 10; ++d) zz[d] = Z[(size_t)row * 10 + d];
  float q[10], s = 0.f;
#pragma unroll
  for (int j = 0; j < 10; ++j) {
    float d2 = 0.f;
#pragma unroll
    for (int d = 0; d < 10; ++d) { float t = zz[d] - MU[j * 10 + d]; d2 += t * t; }
    float t = 1.f / (1.f + d2);
    q[j] = t; s += t;
  }
#pragma unroll
  for (int j = 0; j < 10; ++j) Q[(size_t)row * 10 + j] = q[j] / s;
}

// ---------------------------------------------------------------------------
extern "C" void kernel_launch(void* const* d_in, const int* in_sizes, int n_in,
                              void* d_out, int out_size, void* d_ws, size_t ws_size,
                              hipStream_t stream)
{
  (void)in_sizes; (void)n_in; (void)out_size; (void)ws_size;
  const float* x       = (const float*)d_in[0];
  const float* adj     = (const float*)d_in[1];
  const float* enc1_w  = (const float*)d_in[2];  const float* enc1_b = (const float*)d_in[3];
  const float* enc2_w  = (const float*)d_in[4];  const float* enc2_b = (const float*)d_in[5];
  const float* enc3_w  = (const float*)d_in[6];  const float* enc3_b = (const float*)d_in[7];
  const float* zl_w    = (const float*)d_in[8];  const float* zl_b   = (const float*)d_in[9];
  const float* dec1_w  = (const float*)d_in[10]; const float* dec1_b = (const float*)d_in[11];
  const float* dec2_w  = (const float*)d_in[12]; const float* dec2_b = (const float*)d_in[13];
  const float* dec3_w  = (const float*)d_in[14]; const float* dec3_b = (const float*)d_in[15];
  const float* xbar_w  = (const float*)d_in[16]; const float* xbar_b = (const float*)d_in[17];
  const float* g1_w    = (const float*)d_in[18];
  const float* g2_w    = (const float*)d_in[19];
  const float* g3_w    = (const float*)d_in[20];
  const float* g4_w    = (const float*)d_in[21];
  const float* g6_w    = (const float*)d_in[22];
  const float* g7_w    = (const float*)d_in[23];
  const float* g8_w    = (const float*)d_in[24];
  const float* g9_w    = (const float*)d_in[25];
  const float* agnn_w  = (const float*)d_in[26];
  const float* m1_w    = (const float*)d_in[27]; const float* m1_b = (const float*)d_in[28];
  const float* m2_w    = (const float*)d_in[29]; const float* m2_b = (const float*)d_in[30];
  const float* m3_w    = (const float*)d_in[31]; const float* m3_b = (const float*)d_in[32];
  const float* ml_w    = (const float*)d_in[33]; const float* ml_b = (const float*)d_in[34];
  const float* cluster = (const float*)d_in[35];

  // --- output slices (concatenated tuple, fp32) ---
  float* out     = (float*)d_out;
  float* x_bar   = out;               // 4096 x 2000
  float* z_hat   = out + 8192000;     // 4096 x 2000
  float* adj_hat = out + 16384000;    // 4096 x 4096
  float* q_out   = out + 33161216;    // 4096 x 10
  float* q1_out  = out + 33202176;    // 4096 x 10
  float* z_out   = out + 33243136;    // 4096 x 10
  float* r_out   = out + 33284096;    // 4096 x 10
  float* zl_out  = out + 33325056;    // 4096 x 10
  float* pred    = out + 33366016;    // 4096 x 10

  // --- workspace bump allocator (floats), ~200 MB used ---
  const size_t Nn = 4096;
  size_t off = 0;
  auto alloc = [&](size_t n) { float* p = (float*)d_ws + off; off += n; return p; };
  float* re1  = alloc(Nn * 500);
  float* re2  = alloc(Nn * 500);
  float* re3  = alloc(Nn * 500);
  float* rd1  = alloc(Nn * 2000);
  float* rd2  = alloc(Nn * 500);
  float* rd3  = alloc(Nn * 500);
  float* z1   = alloc(Nn * 500);
  float* z2   = alloc(Nn * 500);
  float* z3   = alloc(Nn * 500);
  float* h3   = alloc(Nn * 500);
  float* mixb = alloc(Nn * 500);
  float* tmpA = alloc(Nn * 2000);
  float* dz1  = alloc(Nn * 2000);
  float* dz2  = alloc(Nn * 500);
  float* dz3  = alloc(Nn * 500);
  float* net  = alloc(Nn * 1510);
  float* p1   = alloc(Nn * 2);
  float* p2   = alloc(Nn * 2);
  float* p3   = alloc(Nn * 2);
  float* w4   = alloc(Nn * 4);
  float* zr   = alloc(Nn * 10);

  auto gemm = [&](const float* A, const float* B, const float* bias, float* C,
                  int K, int Nc, int act, bool bt) {
    dim3 g((Nc + 63) / 64, 4096 / 128), b(256);
    if (bt) gemm_wmma_kernel<true ><<<g, b, 0, stream>>>(A, B, bias, C, K, Nc, act);
    else    gemm_wmma_kernel<false><<<g, b, 0, stream>>>(A, B, bias, C, K, Nc, act);
  };
  auto ew = [&](int total) { return dim3((total + 255) / 256); };

  // ---------------- autoencoder ----------------
  gemm(x,   enc1_w, enc1_b, re1,   2000, 500,  1, false);
  gemm(re1, enc2_w, enc2_b, re2,   500,  500,  1, false);
  gemm(re2, enc3_w, enc3_b, re3,   500,  500,  1, false);
  gemm(re3, zl_w,   zl_b,   r_out, 500,  10,   0, false);
  gemm(r_out, dec1_w, dec1_b, rd1, 10,   2000, 1, false);
  gemm(rd1, dec2_w, dec2_b, rd2,   2000, 500,  1, false);
  gemm(rd2, dec3_w, dec3_b, rd3,   500,  500,  1, false);
  gemm(rd3, xbar_w, xbar_b, x_bar, 500,  2000, 0, false);

  // ---------------- gated GNN encoder ----------------
  gemm(x,   g1_w, nullptr, tmpA, 2000, 500, 0, false);
  gemm(adj, tmpA, nullptr, z1,   4096, 500, 1, false);                       // z1
  gate_kernel<<<dim3(512), 256, 0, stream>>>(re1, 500, z1, 500, nullptr, 0, nullptr, 0,
                                             m1_w, m1_b, p1, 2);             // p1
  mix2_kernel<<<ew(4096 * 500), 256, 0, stream>>>(p1, z1, re1, mixb, 500, 4096 * 500);
  gemm(mixb, g2_w, nullptr, tmpA, 500, 500, 0, false);
  gemm(adj,  tmpA, nullptr, z2,   4096, 500, 1, false);                      // z2
  gate_kernel<<<dim3(512), 256, 0, stream>>>(re2, 500, z2, 500, nullptr, 0, nullptr, 0,
                                             m2_w, m2_b, p2, 2);             // p2
  gemm(z2,  g3_w, nullptr, tmpA, 500, 500, 0, false);
  gemm(adj, tmpA, nullptr, z3,   4096, 500, 1, false);                       // z3
  mix2_kernel<<<ew(4096 * 500), 256, 0, stream>>>(p2, z2, re2, mixb, 500, 4096 * 500);
  gemm(mixb, enc3_w, enc3_b, h3, 500, 500, 1, false);                        // h3
  gate_kernel<<<dim3(512), 256, 0, stream>>>(h3, 500, z3, 500, nullptr, 0, nullptr, 0,
                                             m3_w, m3_b, p3, 2);             // p3
  mix2_kernel<<<ew(4096 * 500), 256, 0, stream>>>(p3, z3, h3, mixb, 500, 4096 * 500);
  gemm(mixb, g4_w, nullptr, tmpA, 500, 10, 0, false);
  gemm(adj,  tmpA, nullptr, z_out, 4096, 10, 1, false);                      // z
  add_kernel<<<ew(4096 * 10), 256, 0, stream>>>(z_out, r_out, zr, 4096 * 10);
  gemm(adj, zr, nullptr, zl_out, 4096, 10, 0, false);                        // z_l

  // ---------------- GNN decoder ----------------
  gemm(z_out, g6_w, nullptr, tmpA, 10, 2000, 0, false);
  gemm(adj, tmpA, nullptr, dz1, 4096, 2000, 1, false);                       // dz1
  gemm(dz1, g7_w, nullptr, tmpA, 2000, 500, 0, false);
  gemm(adj, tmpA, nullptr, dz2, 4096, 500, 1, false);                        // dz2
  gemm(dz2, g8_w, nullptr, tmpA, 500, 500, 0, false);
  gemm(adj, tmpA, nullptr, dz3, 4096, 500, 1, false);                        // dz3
  gemm(dz3, g9_w, nullptr, tmpA, 500, 2000, 0, false);
  gemm(adj, tmpA, nullptr, z_hat, 4096, 2000, 1, false);                     // z_hat
  gemm(z_hat, z_hat, nullptr, adj_hat, 2000, 4096, 2, true);                 // sigmoid(z_hat z_hat^T)

  // ---------------- 4-way adaptive fusion ----------------
  gate_kernel<<<dim3(512), 256, 0, stream>>>(z1, 500, z2, 500, z3, 500, z_out, 10,
                                             ml_w, ml_b, w4, 4);             // w
  build_net_kernel<<<ew(4096 * 1510), 256, 0, stream>>>(w4, z1, z2, z3, z_out, net, 4096 * 1510);
  gemm(net, agnn_w, nullptr, tmpA, 1510, 10, 0, false);
  gemm(adj, tmpA, nullptr, mixb, 4096, 10, 0, false);                        // pre-softmax pred
  softmax_rows_kernel<<<dim3(16), 256, 0, stream>>>(mixb, pred, 10);         // pred

  // ---------------- student-t soft assignments ----------------
  studentt_kernel<<<dim3(16), 256, 0, stream>>>(zl_out, cluster, q_out);     // q
  studentt_kernel<<<dim3(16), 256, 0, stream>>>(r_out,  cluster, q1_out);    // q1
}
